// graphLayer_57389353009178
// MI455X (gfx1250) — compile-verified, matrix operand order checked by reading, not run on hardware
//
#include <hip/hip_runtime.h>
#include <math.h>

typedef __bf16 bf16_t;
typedef bf16_t v16bf __attribute__((ext_vector_type(16)));
typedef float  v8f   __attribute__((ext_vector_type(8)));

#define H2 0.01f  // (0.1)^2

// K-index pattern for 16-bit WMMA A/B fragments (ISA 7.12.2, 16-bit 16x32):
// lanes 0-15: VGPR0..3 -> K=0..7, VGPR4..7 -> K=16..23 ; lanes 16-31: +8
__device__ __forceinline__ int kpat(int i, int hi) {
    int b = (i < 4) ? (2 * i) : (16 + 2 * (i - 4));
    return b + (hi ? 8 : 0);
}

__device__ __forceinline__ void atomAddF32(float* p, float v) {
    unsafeAtomicAdd(p, v);  // hardware global_atomic_add_f32
}

// Branchless fast tanh: 1 - 2/(exp(2x)+1).
// v_exp_f32 + v_rcp_f32, no EXEC divergence (keeps mask all-ones between WMMAs).
// Saturates correctly: exp->+inf => +1, exp->0 => -1. bf16-accuracy is ample.
__device__ __forceinline__ float fast_tanh(float x) {
    float e = __expf(2.0f * x);
    return 1.0f - 2.0f * __builtin_amdgcn_rcpf(e + 1.0f);
}

// ---------------- zero the [N][64] scatter accumulator ----------------
__global__ void __launch_bounds__(256) zero_kernel(float4* __restrict__ p, int n4) {
    int i = blockIdx.x * 256 + threadIdx.x;
    if (i < n4) p[i] = make_float4(0.f, 0.f, 0.f, 0.f);
}

// ---------------- xn [64][N] -> xt [N][64] (LDS-tiled) ----------------
__global__ void __launch_bounds__(256)
transpose_in(const float* __restrict__ xn, float* __restrict__ xt, int NN) {
    __shared__ float tile[64][65];
    const int n0  = blockIdx.x * 64;
    const int tid = threadIdx.x;
    const int nOff = tid & 63;      // fastest across lanes -> coalesced reads
    #pragma unroll
    for (int i = 0; i < 16; ++i) {
        int c = (tid >> 6) + 4 * i;
        int n = n0 + nOff;
        float v = (n < NN) ? xn[(size_t)c * NN + n] : 0.f;
        tile[nOff][c] = v;          // stride-65 rows: bank-conflict free
    }
    __syncthreads();
    const int cOut = tid & 63;      // fastest across lanes -> coalesced writes
    #pragma unroll
    for (int i = 0; i < 16; ++i) {
        int nO = (tid >> 6) + 4 * i;
        int n  = n0 + nO;
        if (n < NN) xt[(size_t)n * 64 + cOut] = tile[nO][cOut];
    }
}

// ---------------- per-edge gather -> WMMA GEMM -> tanh -> WMMA GEMM -> scatter ----------------
__global__ void __launch_bounds__(256)
edge_kernel(const float* __restrict__ xt,
            const int* __restrict__ I, const int* __restrict__ J,
            const float* __restrict__ W, const float* __restrict__ Kf,
            float* __restrict__ dxnT, int E) {
    __shared__ bf16_t Ks[64 * 64];          // A1[m][k] = K[m][k]
    __shared__ bf16_t KTs[64 * 64];         // A2[m][k] = K[k][m]
    __shared__ bf16_t gsh[8][16][64];       // per-wave gradX tile, [edge][ch]
    __shared__ bf16_t ysh[8][16][64];       // per-wave tanh tile,  [edge][ch]

    const int tid = threadIdx.x;
    for (int idx = tid; idx < 64 * 64; idx += 256) {
        float kv = Kf[idx];
        int o = idx >> 6, i = idx & 63;
        Ks[idx]         = (bf16_t)kv;
        KTs[i * 64 + o] = (bf16_t)kv;
    }
    __syncthreads();

    const int wave  = tid >> 5;
    const int lane  = tid & 31;
    const int hi    = lane >> 4;
    const int nl    = lane & 15;
    const int eBase = (blockIdx.x * 8 + wave) * 16;

    bf16_t (*g)[64]  = gsh[wave];
    bf16_t (*ys)[64] = ysh[wave];

    // Gather: one full contiguous 256B column per endpoint per wave iteration.
    for (int e0 = 0; e0 < 16; ++e0) {
        int e = eBase + e0;
        float w = 0.f; int ia = 0, ja = 0;
        if (e < E) { w = W[e]; ia = I[e] * 64; ja = J[e] * 64; }
        float2 xi = *(const float2*)(xt + ia + 2 * lane);
        float2 xj = *(const float2*)(xt + ja + 2 * lane);
        g[e0][2 * lane]     = (bf16_t)(w * (xi.x - xj.x));
        g[e0][2 * lane + 1] = (bf16_t)(w * (xi.y - xj.y));
    }
    // wave-local LDS produce->consume; compiler inserts s_wait_dscnt

    const int eMine = eBase + nl;
    const bool live = (eMine < E);
    float wMine = 0.f; int iaM = 0, jaM = 0;
    if (live) { wMine = W[eMine]; iaM = I[eMine] * 64; jaM = J[eMine] * 64; }

    // GEMM1: y = tanh(K @ g)   (64x64 @ 64x16 per wave)
    #pragma unroll
    for (int mt = 0; mt < 4; ++mt) {
        v8f acc = {0.f, 0.f, 0.f, 0.f, 0.f, 0.f, 0.f, 0.f};
        #pragma unroll
        for (int kc = 0; kc < 2; ++kc) {
            v16bf a, b;
            #pragma unroll
            for (int i = 0; i < 8; ++i) {
                int k = kc * 32 + kpat(i, hi);
                int m = mt * 16 + nl;
                a[2 * i]     = Ks[m * 64 + k];
                a[2 * i + 1] = Ks[m * 64 + k + 1];
                b[2 * i]     = g[nl][k];        // B[k][n] = g[n][k]
                b[2 * i + 1] = g[nl][k + 1];
            }
            acc = __builtin_amdgcn_wmma_f32_16x16x32_bf16(
                false, a, false, b, (short)0, acc, false, false);
        }
        #pragma unroll
        for (int r = 0; r < 8; ++r) {
            int m = mt * 16 + r + 8 * hi;       // C/D layout: row = r + 8*(lane>=16)
            ys[nl][m] = (bf16_t)fast_tanh(acc[r]);
        }
    }

    // GEMM2: dxe = K^T @ y, then scatter +/- W*dxe
    #pragma unroll
    for (int mt = 0; mt < 4; ++mt) {
        v8f acc = {0.f, 0.f, 0.f, 0.f, 0.f, 0.f, 0.f, 0.f};
        #pragma unroll
        for (int kc = 0; kc < 2; ++kc) {
            v16bf a, b;
            #pragma unroll
            for (int i = 0; i < 8; ++i) {
                int k = kc * 32 + kpat(i, hi);
                int m = mt * 16 + nl;
                a[2 * i]     = KTs[m * 64 + k];
                a[2 * i + 1] = KTs[m * 64 + k + 1];
                b[2 * i]     = ys[nl][k];
                b[2 * i + 1] = ys[nl][k + 1];
            }
            acc = __builtin_amdgcn_wmma_f32_16x16x32_bf16(
                false, a, false, b, (short)0, acc, false, false);
        }
        #pragma unroll
        for (int r = 0; r < 8; ++r) {
            int m = mt * 16 + r + 8 * hi;
            float v = wMine * acc[r];
            if (live) {
                atomAddF32(&dxnT[iaM + m],  v);
                atomAddF32(&dxnT[jaM + m], -v);
            }
        }
    }
}

// ---------------- transpose back + leapfrog + second output (copy of xn) ----------------
__global__ void __launch_bounds__(256)
final_kernel(const float* __restrict__ xn, const float* __restrict__ xn_old,
             const float* __restrict__ dxnT, float* __restrict__ out, int NN) {
    __shared__ float tile[64][65];
    const int n0  = blockIdx.x * 64;
    const int tid = threadIdx.x;
    const int cIn = tid & 63;
    #pragma unroll
    for (int i = 0; i < 16; ++i) {
        int nO = (tid >> 6) + 4 * i;
        int n  = n0 + nO;
        tile[nO][cIn] = (n < NN) ? dxnT[(size_t)n * 64 + cIn] : 0.f;
    }
    __syncthreads();
    const int nOff = tid & 63;
    const int n = n0 + nOff;
    #pragma unroll
    for (int i = 0; i < 16; ++i) {
        int c = (tid >> 6) + 4 * i;
        if (n < NN) {
            size_t idx = (size_t)c * NN + n;
            float x = xn[idx];
            out[idx] = 2.f * x - xn_old[idx] - H2 * tile[nOff][c];
            out[(size_t)64 * NN + idx] = x;   // second tuple element = xn
        }
    }
}

extern "C" void kernel_launch(void* const* d_in, const int* in_sizes, int n_in,
                              void* d_out, int out_size, void* d_ws, size_t ws_size,
                              hipStream_t stream) {
    const float* xn     = (const float*)d_in[0];
    const float* xn_old = (const float*)d_in[1];
    const int*   I      = (const int*)d_in[2];
    const int*   J      = (const int*)d_in[3];
    // d_in[4] = N scalar (recovered from in_sizes instead)
    const float* W      = (const float*)d_in[5];
    const float* Kf     = (const float*)d_in[6];
    float* out = (float*)d_out;

    const int NN = in_sizes[0] / 64;   // C = 64 (K is 64x64)
    const int E  = in_sizes[2];

    float* xt   = (float*)d_ws;                  // [NN][64]
    float* dxnT = xt + (size_t)NN * 64;          // [NN][64]

    const int n4 = (NN * 64) / 4;
    zero_kernel<<<(n4 + 255) / 256, 256, 0, stream>>>((float4*)dxnT, n4);
    transpose_in<<<(NN + 63) / 64, 256, 0, stream>>>(xn, xt, NN);
    edge_kernel<<<(E + 127) / 128, 256, 0, stream>>>(xt, I, J, W, Kf, dxnT, E);
    final_kernel<<<(NN + 63) / 64, 256, 0, stream>>>(xn, xn_old, dxnT, out, NN);
}
